// FeatureClusteringMinibatch_45372034515504
// MI455X (gfx1250) — matrix operand order repaired and checked by reading, not compiled, not run on hardware
//
#include <hip/hip_runtime.h>
#include <hip/hip_bf16.h>
#include <math.h>

typedef __attribute__((ext_vector_type(16))) _Float16 v16h;
typedef __attribute__((ext_vector_type(8)))  _Float16 v8h;
typedef __attribute__((ext_vector_type(4)))  _Float16 v4h;
typedef __attribute__((ext_vector_type(8)))  float    v8f;
typedef __attribute__((ext_vector_type(4)))  int      v4i;
typedef __attribute__((address_space(1)))    v4i      v4i_g;   // global
typedef __attribute__((address_space(3)))    v4i      v4i_l;   // LDS

#define NROW  262144
#define CDIM  512
#define NCODE 256
#define BM    64
#define XT_PITCH 520   // halves: 512 + 8 pad (bank-conflict break)
#define VT_PITCH 40    // halves: 32 + 8 pad
#define UV_PITCH 264   // floats: 256 + 8 pad

#if __has_builtin(__builtin_amdgcn_global_load_async_to_lds_b128)
#define ASYNC_LDS 1
#else
#define ASYNC_LDS 0
#endif

__device__ __forceinline__ void wait_async0() {
#if ASYNC_LDS
#if __has_builtin(__builtin_amdgcn_s_wait_asynccnt)
  __builtin_amdgcn_s_wait_asynccnt(0);
#else
  asm volatile("s_wait_asynccnt 0" ::: "memory");
#endif
#endif
}

// Stage one 64B V row (32 f16) for this thread into the LDS tile.
__device__ __forceinline__ void stage_v(const _Float16* __restrict__ Vn16,
                                        _Float16* vt, int t, int kc) {
  const _Float16* src = Vn16 + (size_t)t * CDIM + kc * 32;
  _Float16* dst = vt + t * VT_PITCH;
#if ASYNC_LDS
  #pragma unroll
  for (int i = 0; i < 4; i++)
    __builtin_amdgcn_global_load_async_to_lds_b128(
        (v4i_g*)(void*)(src + i * 8),
        (v4i_l*)(void*)(dst + i * 8), 0, 0);
#else
  const uint4* s4 = (const uint4*)src;
  uint4* d4 = (uint4*)__builtin_assume_aligned(dst, 16);
  uint4 a = s4[0], b = s4[1], c = s4[2], d = s4[3];
  d4[0] = a; d4[1] = b; d4[2] = c; d4[3] = d;
  if (kc + 1 < 16) __builtin_prefetch(src + CDIM, 0, 3);
#endif
}

// ---- WMMA fragment loaders (ISA 7.12.2 layouts, wave32) ----
// A 16x32 f16: lane m=lane&15; halves 0..7 = K k0..k0+7, halves 8..15 = K k0+16..k0+23.
__device__ __forceinline__ v16h frag_a(const _Float16* tile, int pitch, int lane) {
  const int r  = lane & 15;
  const int k0 = (lane >> 4) << 3;
  const _Float16* p = tile + r * pitch + k0;
  v8h lo = *(const v8h*)__builtin_assume_aligned(p, 16);
  v8h hi = *(const v8h*)__builtin_assume_aligned(p + 16, 16);
  return __builtin_shufflevector(lo, hi, 0,1,2,3,4,5,6,7,8,9,10,11,12,13,14,15);
}
// B 32x16 f16: lane n=lane&15; halves 0..7 = K k0..k0+7, halves 8..15 = K k0+16..k0+23, k0 = (lane<16 ? 0 : 8).
__device__ __forceinline__ v16h frag_b(const _Float16* tile, int pitch, int lane) {
  const int r  = lane & 15;
  const int k0 = (lane >> 4) << 3;
  const _Float16* p = tile + r * pitch + k0;
  v8h lo = *(const v8h*)__builtin_assume_aligned(p, 16);
  v8h hi = *(const v8h*)__builtin_assume_aligned(p + 16, 16);
  return __builtin_shufflevector(lo, hi, 0,1,2,3,4,5,6,7,8,9,10,11,12,13,14,15);
}

// ---------------- Kernel 1: normalize codebook ----------------
__global__ __launch_bounds__(256) void k_norm_v(const float* __restrict__ Vb,
                                                float* __restrict__ Vn32,
                                                _Float16* __restrict__ Vn16) {
  const int k = blockIdx.x, t = threadIdx.x;
  float a = Vb[(size_t)k * CDIM + t];
  float b = Vb[(size_t)k * CDIM + t + 256];
  float s = a * a + b * b;
  #pragma unroll
  for (int m = 16; m; m >>= 1) s += __shfl_xor(s, m, 32);
  __shared__ float wsum[8];
  if ((t & 31) == 0) wsum[t >> 5] = s;
  __syncthreads();
  float tot = 0.f;
  #pragma unroll
  for (int i = 0; i < 8; i++) tot += wsum[i];
  const float sc = 1.0f / fmaxf(sqrtf(tot), 1e-12f);
  Vn32[(size_t)k * CDIM + t]       = a * sc;
  Vn32[(size_t)k * CDIM + t + 256] = b * sc;
  Vn16[(size_t)k * CDIM + t]       = (_Float16)(a * sc);
  Vn16[(size_t)k * CDIM + t + 256] = (_Float16)(b * sc);
}

// ---------------- Kernel 2: fused normalize + UV GEMM + softmax stats + csum ----------------
__global__ __launch_bounds__(256) void k_pass1(const float* __restrict__ x,
                                               const _Float16* __restrict__ Vn16,
                                               float* __restrict__ uv_out,
                                               float* __restrict__ rinv_g,
                                               float* __restrict__ rowmax_g,
                                               float* __restrict__ rowinv_g,
                                               float* __restrict__ csum_g) {
  extern __shared__ __align__(16) unsigned char lds[];
  _Float16* xt   = (_Float16*)lds;                   // 64*520*2  = 66560 B
  float*    uvs  = (float*)lds;                      // alias, 64*264*4 = 67584 B
  _Float16* vt0  = (_Float16*)(lds + 67584);         // 256*40*2  = 20480 B
  _Float16* vt1  = (_Float16*)(lds + 88064);         // 20480 B (double buffer)
  float*    rm_s = (float*)(lds + 108544);           // 64 floats
  float*    ri_s = rm_s + 64;                        // 64 floats

  const int t = threadIdx.x, lane = t & 31, wave = t >> 5;
  const int row0 = blockIdx.x * BM;
  const int row4 = t >> 2, part = t & 3;             // 4 threads per row

  // Kick off async DMA of V chunk 0 so it overlaps the normalize phase.
  stage_v(Vn16, vt0, t, 0);

  // Phase 1: load 64x512 f32 tile, L2-normalize rows, stage as f16 in LDS
  {
    const float4* src = (const float4*)(x + (size_t)(row0 + row4) * CDIM + part * 128);
    float s = 0.f;
    #pragma unroll
    for (int i = 0; i < 32; i++) {
      float4 v = src[i];
      s += v.x * v.x + v.y * v.y + v.z * v.z + v.w * v.w;
      v4h h = { (_Float16)v.x, (_Float16)v.y, (_Float16)v.z, (_Float16)v.w };
      *(v4h*)(xt + row4 * XT_PITCH + part * 128 + i * 4) = h;
    }
    s += __shfl_xor(s, 1, 32);
    s += __shfl_xor(s, 2, 32);
    const float sc = 1.0f / fmaxf(sqrtf(s), 1e-12f);
    #pragma unroll
    for (int i = 0; i < 32; i++) {
      v4h h = *(v4h*)(xt + row4 * XT_PITCH + part * 128 + i * 4);
      v4h o = { (_Float16)((float)h[0] * sc), (_Float16)((float)h[1] * sc),
                (_Float16)((float)h[2] * sc), (_Float16)((float)h[3] * sc) };
      *(v4h*)(xt + row4 * XT_PITCH + part * 128 + i * 4) = o;
    }
    if (part == 0) rinv_g[row0 + row4] = sc;
  }
  wait_async0();
  __syncthreads();

  // Phase 2: UV[64x256] = U_tile @ V^T via v_wmma_f32_16x16x32_f16, double-buffered V
  v8f acc[8] = {};
  const int m_tile = wave & 3, nset = wave >> 2;      // wave -> (m, 8 n-tiles)
  for (int kc = 0; kc < 16; kc++) {
    _Float16* cur = (kc & 1) ? vt1 : vt0;
    if (kc + 1 < 16) stage_v(Vn16, (kc & 1) ? vt0 : vt1, t, kc + 1);
    v16h a = frag_a(xt + m_tile * 16 * XT_PITCH + kc * 32, XT_PITCH, lane);
    v16h bfr[8];
    #pragma unroll
    for (int j = 0; j < 8; j++)
      bfr[j] = frag_b(cur + (nset * 8 + j) * 16 * VT_PITCH, VT_PITCH, lane);
    #pragma unroll
    for (int j = 0; j < 8; j++)
      acc[j] = __builtin_amdgcn_wmma_f32_16x16x32_f16(false, a, false, bfr[j],
                                                      (short)0, acc[j], false, false);
    wait_async0();
    __syncthreads();
  }

  // Phase 3: accumulators -> LDS UV tile (aliases dead xt)
  {
    const int rbase = m_tile * 16 + ((lane >> 4) << 3);
    const int col   = lane & 15;
    #pragma unroll
    for (int j = 0; j < 8; j++) {
      const int c = (nset * 8 + j) * 16 + col;
      #pragma unroll
      for (int g = 0; g < 8; g++)
        uvs[(rbase + g) * UV_PITCH + c] = acc[j][g];
    }
  }
  __syncthreads();

  // Phase 4: softmax stats per row (4 lanes/row) + write UV_dist (forward output)
  {
    float vmax = -3.0e38f;
    for (int i = 0; i < 64; i++)
      vmax = fmaxf(vmax, uvs[row4 * UV_PITCH + part * 64 + i]);
    vmax = fmaxf(vmax, __shfl_xor(vmax, 1, 32));
    vmax = fmaxf(vmax, __shfl_xor(vmax, 2, 32));
    float se = 0.f;
    for (int i = 0; i < 64; i++)
      se += __expf(uvs[row4 * UV_PITCH + part * 64 + i] - vmax);
    se += __shfl_xor(se, 1, 32);
    se += __shfl_xor(se, 2, 32);
    const float inv = 1.0f / se;
    if (part == 0) {
      rm_s[row4] = vmax; ri_s[row4] = inv;
      rowmax_g[row0 + row4] = vmax; rowinv_g[row0 + row4] = inv;
    }
    float4* dst = (float4*)(uv_out + (size_t)(row0 + row4) * NCODE + part * 64);
    #pragma unroll
    for (int i = 0; i < 16; i++) {
      float4 v;
      v.x = uvs[row4 * UV_PITCH + part * 64 + i * 4 + 0];
      v.y = uvs[row4 * UV_PITCH + part * 64 + i * 4 + 1];
      v.z = uvs[row4 * UV_PITCH + part * 64 + i * 4 + 2];
      v.w = uvs[row4 * UV_PITCH + part * 64 + i * 4 + 3];
      dst[i] = v;
    }
  }
  __syncthreads();

  // Phase 5: per-block csum contribution (thread = code)
  {
    float s = 0.f;
    for (int r = 0; r < 64; r++)
      s += __expf(uvs[r * UV_PITCH + t] - rm_s[r]) * ri_s[r];
    unsafeAtomicAdd(&csum_g[t], s);
  }
}

// ---------------- Kernel 3: cur_V accumulator = Coeff^T @ U (split-K over N) ----------------
__global__ __launch_bounds__(256) void k_pass2(const float* __restrict__ x,
                                               const float* __restrict__ uv_out,
                                               const float* __restrict__ rinv_g,
                                               const float* __restrict__ rowmax_g,
                                               const float* __restrict__ rowinv_g,
                                               float* __restrict__ curacc) {
  extern __shared__ __align__(16) unsigned char lds[];
  _Float16* uT = (_Float16*)lds;                 // [256 c][32 r] pitch 40 -> 20480 B
  _Float16* aT = (_Float16*)(lds + 20480);       // [64 code][32 r] pitch 40 -> 5120 B

  const int t = threadIdx.x, lane = t & 31, wave = t >> 5;
  const int b  = blockIdx.x;
  const int cg = b & 3;          // code group: 64 codes
  const int ch = (b >> 2) & 1;   // c half: 256 cols
  const int sl = b >> 3;         // N slice: 2048 rows, 128 slices

  const int code_tile = wave & 3, cset = wave >> 2;
  v8f acc[8] = {};

  for (int it = 0; it < 64; it++) {
    const int r = t >> 3, seg = t & 7;
    const size_t grow = (size_t)sl * 2048 + (size_t)it * 32 + r;
    { // stage U^T tile (normalized x -> f16, [c][r])
      const float sc = rinv_g[grow];
      const float4* src = (const float4*)(x + grow * CDIM + ch * 256 + seg * 32);
      if (it + 1 < 64) __builtin_prefetch(x + (grow + 32) * CDIM + ch * 256 + seg * 32, 0, 3);
      #pragma unroll
      for (int i = 0; i < 8; i++) {
        float4 v = src[i];
        const int c = seg * 32 + i * 4;
        uT[(c + 0) * VT_PITCH + r] = (_Float16)(v.x * sc);
        uT[(c + 1) * VT_PITCH + r] = (_Float16)(v.y * sc);
        uT[(c + 2) * VT_PITCH + r] = (_Float16)(v.z * sc);
        uT[(c + 3) * VT_PITCH + r] = (_Float16)(v.w * sc);
      }
    }
    { // stage Coeff^T tile (softmax recomputed from UV + row stats, [code][r])
      const float m = rowmax_g[grow], iv = rowinv_g[grow];
      const float4* src = (const float4*)(uv_out + grow * NCODE + cg * 64 + seg * 8);
      #pragma unroll
      for (int i = 0; i < 2; i++) {
        float4 v = src[i];
        const int k = seg * 8 + i * 4;
        aT[(k + 0) * VT_PITCH + r] = (_Float16)(__expf(v.x - m) * iv);
        aT[(k + 1) * VT_PITCH + r] = (_Float16)(__expf(v.y - m) * iv);
        aT[(k + 2) * VT_PITCH + r] = (_Float16)(__expf(v.z - m) * iv);
        aT[(k + 3) * VT_PITCH + r] = (_Float16)(__expf(v.w - m) * iv);
      }
    }
    __syncthreads();
    v16h a = frag_a(aT + code_tile * 16 * VT_PITCH, VT_PITCH, lane);
    v16h bfr[8];
    #pragma unroll
    for (int j = 0; j < 8; j++)
      bfr[j] = frag_b(uT + (cset * 8 + j) * 16 * VT_PITCH, VT_PITCH, lane);
    #pragma unroll
    for (int j = 0; j < 8; j++)
      acc[j] = __builtin_amdgcn_wmma_f32_16x16x32_f16(false, a, false, bfr[j],
                                                      (short)0, acc[j], false, false);
    __syncthreads();
  }

  // accumulate partial [64 codes][256 c] into global [256][512]
  {
    const int kbase = cg * 64 + code_tile * 16 + ((lane >> 4) << 3);
    const int col   = lane & 15;
    #pragma unroll
    for (int j = 0; j < 8; j++) {
      const int c = ch * 256 + (cset * 8 + j) * 16 + col;
      #pragma unroll
      for (int g = 0; g < 8; g++)
        unsafeAtomicAdd(&curacc[(size_t)(kbase + g) * CDIM + c], acc[j][g]);
    }
  }
}

// ---------------- Kernel 4: EMA finalize ----------------
__global__ __launch_bounds__(256) void k_final(const float* __restrict__ csum_g,
                                               const float* __restrict__ curacc,
                                               const float* __restrict__ Vn32,
                                               const float* __restrict__ Vcnt,
                                               float* __restrict__ out) {
  const int k = blockIdx.x, t = threadIdx.x;
  const float cs = csum_g[k];
  const float vc = Vcnt[k] + cs;
  const float alpha = cs / vc;
  const size_t OFF1 = (size_t)NROW * NCODE;
  #pragma unroll
  for (int q = 0; q < 2; q++) {
    const int c = t + q * 256;
    const float cur = curacc[(size_t)k * CDIM + c] / cs;
    const float vn  = Vn32[(size_t)k * CDIM + c];
    out[OFF1 + (size_t)k * CDIM + c] = (1.f - alpha) * vn + alpha * cur;
  }
  if (t == 0) out[OFF1 + (size_t)NCODE * CDIM + k] = vc;
}

extern "C" void kernel_launch(void* const* d_in, const int* in_sizes, int n_in,
                              void* d_out, int out_size, void* d_ws, size_t ws_size,
                              hipStream_t stream) {
  const float* x  = (const float*)d_in[0];
  const float* Vb = (const float*)d_in[1];
  const float* Vc = (const float*)d_in[2];
  float* out = (float*)d_out;

  char* ws = (char*)d_ws;
  _Float16* Vn16 = (_Float16*)(ws + 0);            // 262144 B
  float* Vn32    = (float*)(ws + 262144);          // 524288 B
  float* curacc  = (float*)(ws + 786432);          // 524288 B
  float* csum    = (float*)(ws + 1310720);         // 1024 B
  float* rinv    = (float*)(ws + 1311744);         // 1 MB
  float* rowmax  = (float*)(ws + 2360320);         // 1 MB
  float* rowinv  = (float*)(ws + 3408896);         // 1 MB

  (void)hipMemsetAsync(curacc, 0, (size_t)NCODE * CDIM * sizeof(float), stream);
  (void)hipMemsetAsync(csum, 0, NCODE * sizeof(float), stream);

  k_norm_v<<<NCODE, 256, 0, stream>>>(Vb, Vn32, Vn16);

  const size_t smem1 = 108544 + 2 * 64 * sizeof(float);
  k_pass1<<<NROW / BM, 256, smem1, stream>>>(x, Vn16, out, rinv, rowmax, rowinv, csum);

  const size_t smem2 = 20480 + 5120;
  k_pass2<<<1024, 256, smem2, stream>>>(x, out, rinv, rowmax, rowinv, curacc);

  k_final<<<NCODE, 256, 0, stream>>>(csum, curacc, Vn32, Vc, out);
}
